// SelfAttention_7421703487580
// MI455X (gfx1250) — compile-verified, matrix operand order checked by reading, not run on hardware
//
#include <hip/hip_runtime.h>

// ---------------------------------------------------------------------------
// Self-attention for MI455X (gfx1250), wave32, WMMA f32_16x16x32_f16.
//   tokens (4096,1024) f32; mask (1,4096) f32; W_Q/K/V (1024,64) f32
//   out = softmax((tokens@Wq)@(tokens@Wk)^T * mask / 8) @ (tokens@Wv)
// Kernel 1: QKV projection (f16 WMMA, f32 accum), Q/K stored f16 row-major,
//           V stored f16 transposed (Vt[64][4096]) for contiguous PV B-frags.
// Kernel 2: fused flash attention, 16 query rows per wave, 128-key blocks
//           staged in LDS (K block via Tensor Data Mover with LDS row
//           padding; Vt via b128 loads), online softmax, all GEMMs via wmma.
// ---------------------------------------------------------------------------

typedef __attribute__((ext_vector_type(16))) _Float16 v16h;
typedef __attribute__((ext_vector_type(8)))  float    v8f;
typedef __attribute__((ext_vector_type(4)))  unsigned v4u;
typedef __attribute__((ext_vector_type(8)))  int      v8i;
typedef __attribute__((ext_vector_type(4)))  int      v4i;

#define WMMA_F16(a, b, c) \
  __builtin_amdgcn_wmma_f32_16x16x32_f16(false, (a), false, (b), (short)0, (c), false, false)

#if defined(__has_builtin)
#if __has_builtin(__builtin_amdgcn_tensor_load_to_lds) && \
    __has_builtin(__builtin_amdgcn_s_wait_tensorcnt)
#define USE_TDM 1
#endif
#endif

union FragH {
  v16h v;
  _Float16 h[16];
  uint4 q[2];
};

// Layout constants
#define NTOK   4096
#define EMBED  1024
#define ATTN   64
// kernel1 LDS: W chunk transposed, 64 dims x 256 k, stride padded to 264 halves
#define WSTR   264
#define KCHUNK 256
// kernel2 LDS strides (halves), padded for bank spread, 16B aligned
#define BK     128
#define KSTR   72      // 64 + 8: matches TDM pad (4 dwords every 32 dwords)
#define VSTR   136
#define PSTR   136

// ---------------------------------------------------------------------------
// Kernel 1: C(4096x64) = tokens(4096x1024) @ W(1024x64), one of Q/K/V per
// blockIdx.y. 8 waves/block, 16 rows/wave -> 128 rows/block, 32 blocks in x.
// ---------------------------------------------------------------------------
__global__ __launch_bounds__(256) void qkv_proj_kernel(
    const float* __restrict__ tokens,
    const float* __restrict__ wq,
    const float* __restrict__ wk,
    const float* __restrict__ wv,
    _Float16* __restrict__ qh,     // [4096][64]
    _Float16* __restrict__ kh,     // [4096][64]
    _Float16* __restrict__ vth)    // [64][4096]  (transposed V)
{
  __shared__ _Float16 wt[ATTN * WSTR];   // ~33.8 KB

  const int tid    = threadIdx.x;
  const int wave   = tid >> 5;
  const int lane   = tid & 31;
  const int hlf    = lane >> 4;          // 0 or 1 (which 16-lane half)
  const int l16    = lane & 15;
  const int mat    = blockIdx.y;         // 0=Q 1=K 2=V
  const float* W   = (mat == 0) ? wq : (mat == 1) ? wk : wv;
  const int row0   = blockIdx.x * 128 + wave * 16;

  v8f acc[4] = {};                       // 16 rows x 64 cols accumulators

  for (int c = 0; c < EMBED / KCHUNK; ++c) {
    const int k0 = c * KCHUNK;
    __syncthreads();
    // Stage W chunk transposed into LDS as f16: wt[n][kk] = W[k0+kk][n].
    // float4 per thread: coalesced 16B global reads, 16 passes.
    for (int p = 0; p < (ATTN * KCHUNK) / (256 * 4); ++p) {
      int u  = tid + p * 256;
      int n4 = (u & 15) * 4;
      int kk = u >> 4;
      float4 w4 = *(const float4*)(W + (size_t)(k0 + kk) * ATTN + n4);
      wt[(n4 + 0) * WSTR + kk] = (_Float16)w4.x;
      wt[(n4 + 1) * WSTR + kk] = (_Float16)w4.y;
      wt[(n4 + 2) * WSTR + kk] = (_Float16)w4.z;
      wt[(n4 + 3) * WSTR + kk] = (_Float16)w4.w;
    }
    __syncthreads();

    #pragma unroll
    for (int ks = 0; ks < KCHUNK / 32; ++ks) {
      const int kb = ks * 32;
      // Batch all 4 B fragments first (8 ds_load_b128 in one clause),
      // then the A fragment, then 4 back-to-back WMMAs behind one wait.
      FragH b[4];
      #pragma unroll
      for (int t = 0; t < 4; ++t) {
        const _Float16* bp = wt + (t * 16 + l16) * WSTR + kb + hlf * 8;
        b[t].q[0] = *(const uint4*)(bp);
        b[t].q[1] = *(const uint4*)(bp + 16);
      }
      // A fragment: tokens row (lane&15), k = kb + {g*16 + hlf*8 + j}
      FragH a;
      {
        const float* ar = tokens + (size_t)(row0 + l16) * EMBED + k0 + kb + hlf * 8;
        float af[16];
        *(float4*)(af + 0)  = *(const float4*)(ar);
        *(float4*)(af + 4)  = *(const float4*)(ar + 4);
        *(float4*)(af + 8)  = *(const float4*)(ar + 16);
        *(float4*)(af + 12) = *(const float4*)(ar + 20);
        #pragma unroll
        for (int i = 0; i < 16; ++i) a.h[i] = (_Float16)af[i];
      }
      #pragma unroll
      for (int t = 0; t < 4; ++t) acc[t] = WMMA_F16(a.v, b[t].v, acc[t]);
    }
    __syncthreads();
  }

  // C-layout: vgpr r holds row r (lanes 0-15) / row r+8 (lanes 16-31), col=lane&15
  #pragma unroll
  for (int t = 0; t < 4; ++t) {
    #pragma unroll
    for (int r = 0; r < 8; ++r) {
      const int row = row0 + r + 8 * hlf;
      const int col = t * 16 + l16;
      const _Float16 hv = (_Float16)acc[t][r];
      if (mat == 2)      vth[(size_t)col * NTOK + row] = hv;
      else if (mat == 1) kh[(size_t)row * ATTN + col]  = hv;
      else               qh[(size_t)row * ATTN + col]  = hv;
    }
  }
}

// ---------------------------------------------------------------------------
// Kernel 2: fused flash attention. 4 waves/block, 16 query rows per wave,
// loop over 128-key blocks staged in LDS.
// ---------------------------------------------------------------------------
__global__ __launch_bounds__(128) void flash_attn_kernel(
    const _Float16* __restrict__ qh,
    const _Float16* __restrict__ kh,
    const _Float16* __restrict__ vth,
    const float* __restrict__ mask,
    float* __restrict__ out)
{
  __shared__ _Float16 kl[BK * KSTR];        // K block  [key][dim]  18.4 KB
  __shared__ _Float16 vl[ATTN * VSTR];      // Vt block [dim][key]  17.4 KB
  __shared__ _Float16 pl[4 * 16 * PSTR];    // per-wave P staging   17.4 KB

  const int tid  = threadIdx.x;
  const int wave = tid >> 5;
  const int lane = tid & 31;
  const int hlf  = lane >> 4;
  const int l16  = lane & 15;
  const int q0   = blockIdx.x * 64 + wave * 16;

  // Q fragments (16 rows x 64 dims -> 2 A-frags), from f16 workspace
  FragH qf[2];
  {
    const _Float16* qr = qh + (size_t)(q0 + l16) * ATTN;
    #pragma unroll
    for (int kq = 0; kq < 2; ++kq) {
      const _Float16* p = qr + kq * 32 + hlf * 8;
      qf[kq].q[0] = *(const uint4*)(p);
      qf[kq].q[1] = *(const uint4*)(p + 16);
    }
  }

  float m[8], l[8];
  v8f oacc[4] = {};
  #pragma unroll
  for (int r = 0; r < 8; ++r) { m[r] = -__builtin_inff(); l[r] = 0.0f; }

  _Float16* pw = pl + wave * 16 * PSTR;
  const float scale = 0.125f;               // 1/sqrt(64)

  for (int j = 0; j < NTOK / BK; ++j) {
    const int key0 = j * BK;
    __syncthreads();

#ifdef USE_TDM
    // Stage K block with the Tensor Data Mover: 128x64 f16 tile of the
    // [4096][64] tensor, LDS padded 4 dwords per 32 dwords (-> KSTR=72).
    // TDM ignores EXEC, so gate with a uniform scalar branch: wave 0 only.
    if (wave == 0) {
      const unsigned lds_kl = (unsigned)(size_t)(const void*)kl;
      const unsigned long long ga =
          (unsigned long long)(size_t)kh + (unsigned long long)key0 * (ATTN * 2);
      // D# group0: count=1 | lds_addr | global_addr[56:0] | type=2
      v4u g0 = { 1u,
                 lds_kl,
                 (unsigned)(ga & 0xffffffffull),
                 (unsigned)((ga >> 32) & 0x01ffffffull) | (2u << 30) };
      // D# group1: data_size=2B, pad_enable, pad_interval=32dw(code 4),
      // pad_amount=4dw(code 3); tensor 64 x 4096, tile 64 x 128, stride 64.
      v8i g1 = { (int)((1u << 16) | (1u << 20) | (4u << 22) | (3u << 25)),
                 (int)(64u << 16),      // tensor_dim0 = 64      (bits 79:48)
                 (int)(4096u << 16),    // tensor_dim1 = 4096    (bits 111:80)
                 (int)(64u << 16),      // tile_dim0 = 64        (bits 127:112)
                 (int)128,              // tile_dim1 = 128       (bits 143:128)
                 (int)64,               // tensor_dim0_stride=64 (bits 207:160)
                 0, 0 };
      v4i gz4 = { 0, 0, 0, 0 };                  // groups 2/3 unused (2D tile)
      v8i gz8 = { 0, 0, 0, 0, 0, 0, 0, 0 };      // trailing group (clang-23 form)
      __builtin_amdgcn_tensor_load_to_lds(g0, g1, gz4, gz4, gz8, 0);
      __builtin_amdgcn_s_wait_tensorcnt(0);
    }
#else
    // Fallback: stage K block with b128 loads: kl[key][d] = kh[key0+key][d]
    #pragma unroll
    for (int p = 0; p < 8; ++p) {
      int u = tid + p * 128;
      int key = u >> 3, d8 = u & 7;
      *(uint4*)(kl + key * KSTR + d8 * 8) =
          *(const uint4*)(kh + (size_t)(key0 + key) * ATTN + d8 * 8);
    }
#endif
    // Stage Vt block: vl[d][key] = vth[d][key0+key]
    #pragma unroll
    for (int p = 0; p < 8; ++p) {
      int u = tid + p * 128;
      int dim = u >> 4, k8 = u & 15;
      *(uint4*)(vl + dim * VSTR + k8 * 8) =
          *(const uint4*)(vth + (size_t)dim * NTOK + key0 + k8 * 8);
    }
    __syncthreads();

    // scores: 8 tiles of 16 keys; process tiles in pairs so 8 ds_load_b128
    // clause together and 4 WMMAs issue behind a single dscnt wait.
    v8f s[8];
    #pragma unroll
    for (int tt = 0; tt < 8; tt += 2) {
      FragH b[4];
      #pragma unroll
      for (int i = 0; i < 2; ++i) {
        const _Float16* kp = kl + ((tt + i) * 16 + l16) * KSTR + hlf * 8;
        b[2 * i + 0].q[0] = *(const uint4*)(kp);
        b[2 * i + 0].q[1] = *(const uint4*)(kp + 16);
        b[2 * i + 1].q[0] = *(const uint4*)(kp + 32);
        b[2 * i + 1].q[1] = *(const uint4*)(kp + 48);
      }
      v8f c0 = {};
      c0 = WMMA_F16(qf[0].v, b[0].v, c0);
      c0 = WMMA_F16(qf[1].v, b[1].v, c0);
      s[tt] = c0;
      v8f c1 = {};
      c1 = WMMA_F16(qf[0].v, b[2].v, c1);
      c1 = WMMA_F16(qf[1].v, b[3].v, c1);
      s[tt + 1] = c1;
    }

    // multiplicative mask + scale, per-lane row-max across the 8 tiles
    float mt[8];
    #pragma unroll
    for (int r = 0; r < 8; ++r) mt[r] = -__builtin_inff();
    #pragma unroll
    for (int t = 0; t < 8; ++t) {
      const float mv = mask[key0 + t * 16 + l16] * scale;
      #pragma unroll
      for (int r = 0; r < 8; ++r) {
        float v = s[t][r] * mv;
        s[t][r] = v;
        mt[r] = fmaxf(mt[r], v);
      }
    }
    // row-max across the 16 lanes holding each row (xor 1,2,4,8 stays in half)
    #pragma unroll
    for (int r = 0; r < 8; ++r) {
      float v = mt[r];
      v = fmaxf(v, __shfl_xor(v, 1, 32));
      v = fmaxf(v, __shfl_xor(v, 2, 32));
      v = fmaxf(v, __shfl_xor(v, 4, 32));
      v = fmaxf(v, __shfl_xor(v, 8, 32));
      mt[r] = v;
    }
    float alpha[8];
    #pragma unroll
    for (int r = 0; r < 8; ++r) {
      float mnew = fmaxf(m[r], mt[r]);
      alpha[r] = __expf(m[r] - mnew);
      m[r] = mnew;
      l[r] *= alpha[r];
    }
    #pragma unroll
    for (int t = 0; t < 4; ++t)
      #pragma unroll
      for (int r = 0; r < 8; ++r) oacc[t][r] *= alpha[r];

    // P = exp(s - m): accumulate lane-partial row sums, stage f16 in LDS
    #pragma unroll
    for (int t = 0; t < 8; ++t) {
      #pragma unroll
      for (int r = 0; r < 8; ++r) {
        float p = __expf(s[t][r] - m[r]);
        l[r] += p;
        pw[(r + 8 * hlf) * PSTR + t * 16 + l16] = (_Float16)p;
      }
    }

    // out += P(16x128) @ V(128x64): 4 k-steps x 4 dim-tiles, loads batched
    #pragma unroll
    for (int kt = 0; kt < 4; ++kt) {
      FragH a, b[4];
      const _Float16* ap = pw + l16 * PSTR + kt * 32 + hlf * 8;
      a.q[0] = *(const uint4*)(ap);
      a.q[1] = *(const uint4*)(ap + 16);
      #pragma unroll
      for (int t = 0; t < 4; ++t) {
        const _Float16* vp = vl + (t * 16 + l16) * VSTR + kt * 32 + hlf * 8;
        b[t].q[0] = *(const uint4*)(vp);
        b[t].q[1] = *(const uint4*)(vp + 16);
      }
      #pragma unroll
      for (int t = 0; t < 4; ++t) oacc[t] = WMMA_F16(a.v, b[t].v, oacc[t]);
    }
  }

  // finalize: reduce lane-partial row sums, normalize, store f32
  #pragma unroll
  for (int r = 0; r < 8; ++r) {
    float v = l[r];
    v += __shfl_xor(v, 1, 32);
    v += __shfl_xor(v, 2, 32);
    v += __shfl_xor(v, 4, 32);
    v += __shfl_xor(v, 8, 32);
    l[r] = 1.0f / v;
  }
  #pragma unroll
  for (int t = 0; t < 4; ++t)
    #pragma unroll
    for (int r = 0; r < 8; ++r) {
      const int row = q0 + r + 8 * hlf;
      out[(size_t)row * ATTN + t * 16 + l16] = oacc[t][r] * l[r];
    }
}

// ---------------------------------------------------------------------------
extern "C" void kernel_launch(void* const* d_in, const int* in_sizes, int n_in,
                              void* d_out, int out_size, void* d_ws, size_t ws_size,
                              hipStream_t stream) {
  (void)in_sizes; (void)n_in; (void)out_size; (void)ws_size;
  const float* tokens = (const float*)d_in[0];
  const float* mask   = (const float*)d_in[1];
  const float* wq     = (const float*)d_in[2];
  const float* wk     = (const float*)d_in[3];
  const float* wv     = (const float*)d_in[4];
  float* out = (float*)d_out;

  // workspace: Q, K (row-major f16) and V transposed (f16) — 1.5 MB total
  _Float16* qh = (_Float16*)d_ws;
  _Float16* kh = qh + (size_t)NTOK * ATTN;
  _Float16* vt = kh + (size_t)NTOK * ATTN;

  dim3 g1(NTOK / 128, 3), b1(256);
  qkv_proj_kernel<<<g1, b1, 0, stream>>>(tokens, wq, wk, wv, qh, kh, vt);
  flash_attn_kernel<<<NTOK / 64, 128, 0, stream>>>(qh, kh, vt, mask, out);
}
